// Fwhat_66108136620386
// MI455X (gfx1250) — compile-verified, matrix operand order checked by reading, not run on hardware
//
#include <hip/hip_runtime.h>

typedef __attribute__((ext_vector_type(16))) _Float16 v16h;
typedef __attribute__((ext_vector_type(8)))  float    v8f;

#define B_  256
#define T_  2048
#define X_  64
#define H_  128
#define Y_  32
#define MT  32   // batch rows per workgroup -> 8 workgroups

__device__ __forceinline__ v8f wmma16(v16h a, v16h b, v8f c) {
  // D = A(16x32 f16) * B(32x16 f16) + C(16x16 f32)
  return __builtin_amdgcn_wmma_f32_16x16x32_f16(false, a, false, b, (short)0, c, false, false);
}

// load 16 contiguous f32 from global, convert to packed f16 fragment
__device__ __forceinline__ v16h load16_cvt(const float* p) {
  const float4* p4 = (const float4*)p;
  v16h r;
#pragma unroll
  for (int i = 0; i < 4; ++i) {
    float4 v = p4[i];
    r[4*i+0] = (_Float16)v.x;
    r[4*i+1] = (_Float16)v.y;
    r[4*i+2] = (_Float16)v.z;
    r[4*i+3] = (_Float16)v.w;
  }
  return r;
}

__device__ __forceinline__ float fast_tanh(float x) {
  // tanh(x) = 1 - 2/(exp(2x)+1);  exp(2x) = exp2(x * 2/ln2)
  float e = __builtin_amdgcn_exp2f(x * 2.885390081777927f);
  return 1.0f - 2.0f * __builtin_amdgcn_rcpf(e + 1.0f);
}

__global__ __launch_bounds__(256, 1)
void rnn_scan_kernel(const float* __restrict__ x,
                     const float* __restrict__ W_ih,
                     const float* __restrict__ W_hh,
                     const float* __restrict__ b_ih,
                     const float* __restrict__ b_hh,
                     const float* __restrict__ W_out,
                     const float* __restrict__ b_out,
                     float* __restrict__ out) {
  // double-buffered hidden state, f16, [32 rows x 128 cols] each
  __shared__ __attribute__((aligned(32))) _Float16 hbuf[2][MT * H_];

  const int tid   = threadIdx.x;
  const int wave  = tid >> 5;     // 8 waves; wave w owns output cols [16w, 16w+16)
  const int lane  = tid & 31;
  const int lhalf = lane >> 4;    // K-half select for A/B fragments
  const int lmod  = lane & 15;    // row (A) / column (B,C,D) within tile
  const int n0    = wave * 16;
  const int b0    = blockIdx.x * MT;

  // ---- persistent weight B-fragments in VGPRs (column n = n0+lmod) ----
  v16h Bhh[4];                    // W_hh: K = 128 -> 4 chunks of 32
#pragma unroll
  for (int kt = 0; kt < 4; ++kt)
    Bhh[kt] = load16_cvt(W_hh + (size_t)(n0 + lmod) * H_ + 32*kt + 16*lhalf);
  v16h Bih[2];                    // W_ih: K = 64 -> 2 chunks of 32
#pragma unroll
  for (int kt = 0; kt < 2; ++kt)
    Bih[kt] = load16_cvt(W_ih + (size_t)(n0 + lmod) * X_ + 32*kt + 16*lhalf);

  const float bias = b_ih[n0 + lmod] + b_hh[n0 + lmod];

  // zero h0 (buffer 0): 8192 bytes, 32 B per thread
  {
    uint4* z = (uint4*)&hbuf[0][0];
    z[tid]       = make_uint4(0u, 0u, 0u, 0u);
    z[tid + 256] = make_uint4(0u, 0u, 0u, 0u);
  }
  __syncthreads();

#pragma unroll 1
  for (int t = 0; t < T_; ++t) {
    const int rb = t & 1;
    const int wb = rb ^ 1;

    // x-tile A fragments for this timestep (f32 -> f16), prefetch next step
    v16h Ax[2][2];
#pragma unroll
    for (int mt = 0; mt < 2; ++mt)
#pragma unroll
      for (int kt = 0; kt < 2; ++kt) {
        const float* px = x + ((size_t)(b0 + 16*mt + lmod) * T_ + t) * X_
                            + 32*kt + 16*lhalf;
        Ax[mt][kt] = load16_cvt(px);
        if (t + 1 < T_) __builtin_prefetch(px + X_, 0, 0);  // global_prefetch_b8
      }

    v8f c0, c1;
#pragma unroll
    for (int r = 0; r < 8; ++r) { c0[r] = bias; c1[r] = bias; }

    // input projection: c += x_t * W_ih^T   (interleave M-tiles to hide RAW)
    c0 = wmma16(Ax[0][0], Bih[0], c0);
    c1 = wmma16(Ax[1][0], Bih[0], c1);
    c0 = wmma16(Ax[0][1], Bih[1], c0);
    c1 = wmma16(Ax[1][1], Bih[1], c1);

    // recurrence: c += h_{t-1} * W_hh^T   (A from LDS f16)
#pragma unroll
    for (int kt = 0; kt < 4; ++kt) {
      v16h a0 = *(const v16h*)&hbuf[rb][(lmod)      * H_ + 32*kt + 16*lhalf];
      v16h a1 = *(const v16h*)&hbuf[rb][(16 + lmod) * H_ + 32*kt + 16*lhalf];
      c0 = wmma16(a0, Bhh[kt], c0);
      c1 = wmma16(a1, Bhh[kt], c1);
    }

    // tanh, convert to f16, write to the other LDS buffer
#pragma unroll
    for (int r = 0; r < 8; ++r) {
      float h0 = fast_tanh(c0[r]);
      float h1 = fast_tanh(c1[r]);
      hbuf[wb][(r + 8*lhalf)      * H_ + n0 + lmod] = (_Float16)h0;
      hbuf[wb][(16 + r + 8*lhalf) * H_ + n0 + lmod] = (_Float16)h1;
    }
    __syncthreads();   // write(t)->read(t+1) and read(t)->write(t+1) hazards
  }

  // ---- readout: out = h_last @ W_out^T + b_out ; h_last is in hbuf[0] ----
  if (wave < 2) {      // Y = 32 -> waves 0,1 cover all output columns
    v16h Bo[4];
#pragma unroll
    for (int kt = 0; kt < 4; ++kt)
      Bo[kt] = load16_cvt(W_out + (size_t)(n0 + lmod) * H_ + 32*kt + 16*lhalf);
    const float bo = b_out[n0 + lmod];
    v8f c0, c1;
#pragma unroll
    for (int r = 0; r < 8; ++r) { c0[r] = bo; c1[r] = bo; }
#pragma unroll
    for (int kt = 0; kt < 4; ++kt) {
      v16h a0 = *(const v16h*)&hbuf[0][(lmod)      * H_ + 32*kt + 16*lhalf];
      v16h a1 = *(const v16h*)&hbuf[0][(16 + lmod) * H_ + 32*kt + 16*lhalf];
      c0 = wmma16(a0, Bo[kt], c0);
      c1 = wmma16(a1, Bo[kt], c1);
    }
#pragma unroll
    for (int r = 0; r < 8; ++r) {
      out[(size_t)(b0 + r + 8*lhalf)      * Y_ + n0 + lmod] = c0[r];
      out[(size_t)(b0 + 16 + r + 8*lhalf) * Y_ + n0 + lmod] = c1[r];
    }
  }
}

extern "C" void kernel_launch(void* const* d_in, const int* in_sizes, int n_in,
                              void* d_out, int out_size, void* d_ws, size_t ws_size,
                              hipStream_t stream) {
  (void)in_sizes; (void)n_in; (void)out_size; (void)d_ws; (void)ws_size;
  const float* x     = (const float*)d_in[0];
  const float* W_ih  = (const float*)d_in[1];
  const float* W_hh  = (const float*)d_in[2];
  const float* b_ih  = (const float*)d_in[3];
  const float* b_hh  = (const float*)d_in[4];
  const float* W_out = (const float*)d_in[5];
  const float* b_out = (const float*)d_in[6];
  float* out = (float*)d_out;

  dim3 grid(B_ / MT);   // 8 workgroups (one per 32-row batch tile)
  dim3 block(256);      // 8 wave32 waves
  rnn_scan_kernel<<<grid, block, 0, stream>>>(x, W_ih, W_hh, b_ih, b_hh,
                                              W_out, b_out, out);
}
